// C_eAR_GenRNN_13709535609052
// MI455X (gfx1250) — matrix operand order repaired and checked by reading, not compiled, not run on hardware
//
#include <hip/hip_runtime.h>
#include <hip/hip_bf16.h>

// Problem dims
#define B_    32
#define T_    4096
#define EMB_  256
#define ICND_ 512
#define DIN_  768
#define H_    896
#define G3_   2688     // 3*H
#define O_    512
#define NWG_  56       // recurrence workgroups (H/16)

typedef __attribute__((ext_vector_type(16))) __bf16 v16bf;
typedef __attribute__((ext_vector_type(8)))  float  v8f;
typedef __attribute__((ext_vector_type(4)))  unsigned int u32x4;

union Frag { u32x4 q[2]; v16bf v; };

__device__ __forceinline__ unsigned short f2bf(float f) {
  unsigned int u = __float_as_uint(f);
  unsigned int r = u + 0x7FFFu + ((u >> 16) & 1u);   // round-to-nearest-even
  return (unsigned short)(r >> 16);
}
__device__ __forceinline__ float bf2f(unsigned short h) {
  return __uint_as_float(((unsigned int)h) << 16);
}
// fast gate math: v_exp_f32 + raw v_rcp_f32 (no IEEE divide chain)
__device__ __forceinline__ float fast_sigmoid(float x) {
  return __builtin_amdgcn_rcpf(1.f + __expf(-x));
}
__device__ __forceinline__ float fast_tanh(float x) {
  return 1.f - 2.f * __builtin_amdgcn_rcpf(__expf(2.f * x) + 1.f);
}

// ---------------------------------------------------------------------------
// Pack a row-major (N,K) f32 weight into bf16 WMMA B-operand tiles.
// Tile (ntile,ktile) = 16 cols x 32 K. Lane l holds col n = ntile*16+(l&15),
// K run = ktile*32 + 16*(l>>4) .. +15. 512 bf16 per tile, lane-major
// (16 elems = 32B per lane) so a wave loads a tile with 2x b128.
// ---------------------------------------------------------------------------
__global__ void pack_w(const float* __restrict__ w, unsigned short* __restrict__ out,
                       int N, int K, int total) {
  int e = blockIdx.x * blockDim.x + threadIdx.x;
  if (e >= total) return;
  int tile = e >> 9, within = e & 511;
  int lane = within >> 4, j = within & 15;
  int ktiles = K >> 5;
  int ntile = tile / ktiles, ktile = tile - ntile * ktiles;
  int n = ntile * 16 + (lane & 15);
  int k = ktile * 32 + ((lane >> 4) << 4) + j;
  out[e] = f2bf(w[(size_t)n * K + k]);
}

// x = concat(emb[idx], cnd) gathered into (T*B, 768) bf16, token order t*B+b
__global__ void gather_xb(const int* __restrict__ idx, const float* __restrict__ cnd,
                          const float* __restrict__ emb, unsigned short* __restrict__ xb,
                          int total) {
  int e = blockIdx.x * blockDim.x + threadIdx.x;
  if (e >= total) return;
  int row = e / DIN_, col = e - row * DIN_;
  int t = row >> 5, b = row & 31;
  float v;
  if (col < EMB_) v = emb[(size_t)idx[b * T_ + t] * EMB_ + col];
  else            v = cnd[((size_t)b * T_ + t) * ICND_ + (col - EMB_)];
  xb[e] = f2bf(v);
}

__global__ void zero_init(unsigned short* hbuf, int nh, int* ctrs, int nc) {
  int e = blockIdx.x * blockDim.x + threadIdx.x;
  if (e < nh) hbuf[e] = 0;
  if (e < nc) ctrs[e] = 0;
}

// ---------------------------------------------------------------------------
// Generic bf16 WMMA GEMM: C[M,N] = A[M,K] * Bp^T + bias. Block = 256 thr
// (8 waves, 4Mx2N), block tile 128x128, wave tile 32x64 (2x4 C tiles).
// EPI: 0=store bf16, 1=relu+bf16, 2=f32 with (t*B+b)->(b,T,t) permuted store.
// ---------------------------------------------------------------------------
template<int EPI, typename OutT>
__global__ __launch_bounds__(256) void gemm_bf16(
    const unsigned short* __restrict__ A, const unsigned short* __restrict__ Bp,
    const float* __restrict__ bias, OutT* __restrict__ C, int M, int N, int K) {
  const int lane = threadIdx.x & 31;
  const int wave = threadIdx.x >> 5;
  const int wm = wave >> 1, wn = wave & 1;
  const int rowBase = blockIdx.x * 128 + wm * 32;
  const int colBase = blockIdx.y * 128 + wn * 64;
  const int ktiles = K >> 5;

  v8f acc[2][4];
#pragma unroll
  for (int i = 0; i < 2; ++i)
#pragma unroll
    for (int j = 0; j < 4; ++j) { v8f z = {}; acc[i][j] = z; }

  for (int kt = 0; kt < ktiles; ++kt) {
    Frag a[2], b[4];
    const int run1 = (kt << 5) + ((lane >> 4) << 4);  // bf16 elements
#pragma unroll
    for (int i = 0; i < 2; ++i) {
      int m = rowBase + i * 16 + (lane & 15);
      const unsigned short* base = A + (size_t)m * K + run1;
      a[i].q[0] = *reinterpret_cast<const u32x4*>(base);
      a[i].q[1] = *reinterpret_cast<const u32x4*>(base + 8);
    }
#pragma unroll
    for (int j = 0; j < 4; ++j) {
      int ntile = (colBase + j * 16) >> 4;
      const unsigned short* base = Bp + (((size_t)(ntile * ktiles + kt)) << 9) + (lane << 4);
      b[j].q[0] = *reinterpret_cast<const u32x4*>(base);
      b[j].q[1] = *reinterpret_cast<const u32x4*>(base + 8);
    }
    if (kt + 1 < ktiles) {
      int ntile0 = colBase >> 4;
      __builtin_prefetch(Bp + (((size_t)(ntile0 * ktiles + kt + 1)) << 9), 0, 1);
    }
#pragma unroll
    for (int i = 0; i < 2; ++i)
#pragma unroll
      for (int j = 0; j < 4; ++j)
        acc[i][j] = __builtin_amdgcn_wmma_f32_16x16x32_bf16(
            false, a[i].v, false, b[j].v, (short)0, acc[i][j], false, false);
  }

#pragma unroll
  for (int i = 0; i < 2; ++i) {
#pragma unroll
    for (int j = 0; j < 4; ++j) {
      int n = colBase + j * 16 + (lane & 15);
      float bv = bias[n];
#pragma unroll
      for (int v = 0; v < 8; ++v) {
        int m = rowBase + i * 16 + ((lane >> 4) << 3) + v;
        float val = acc[i][j][v] + bv;
        if constexpr (EPI == 1) val = fmaxf(val, 0.f);
        if constexpr (EPI == 2) {
          int b_ = m & 31, t_ = m >> 5;               // m = t*B + b
          C[((size_t)b_ * T_ + t_) * O_ + n] = val;   // (B,T,O) output
        } else {
          C[(size_t)m * N + n] = (OutT)f2bf(val);
        }
      }
    }
  }
}

// ---------------------------------------------------------------------------
// GRU recurrence. 56 WGs x 192 thr (6 waves). WG c owns h dims [16c,16c+16):
// its 48 w_hh rows (r/z/n) live in LDS as packed B tiles (84 KB, loaded once).
// Per step: gx loads hoisted to overlap the 28-WMMA gh GEMM; gate math in
// fp32 (straight-line, fixed 3 slots/thread); h slice broadcast via L2;
// full h pulled back with global_load_async_to_lds_b128 (ASYNCcnt).
// ---------------------------------------------------------------------------
__device__ __forceinline__ void grid_sync(int* bar) {
  __syncthreads();
  if (threadIdx.x == 0) {
    volatile int* vgen = bar + 1;
    int g = *vgen;
    __threadfence();
    unsigned int arrived = atomicAdd((unsigned int*)&bar[0], 1u);
    if (arrived == (unsigned int)(NWG_ - 1)) {
      bar[0] = 0;
      __threadfence();
      *vgen = g + 1;
    } else {
      while (*vgen == g) __builtin_amdgcn_s_sleep(1);
    }
    __threadfence();
  }
  __syncthreads();
}

__global__ __launch_bounds__(192) void gru_scan(
    const unsigned short* __restrict__ whhp,   // packed w_hh B tiles
    const unsigned short* __restrict__ gates,  // (T*B, 3H) bf16, incl b_ih
    const float* __restrict__ bhh,             // (3H)
    unsigned short* __restrict__ hbuf,         // (B, H) bf16 broadcast buffer
    unsigned short* __restrict__ ornn,         // (T*B, H) bf16 GRU output
    int* __restrict__ bar) {
  extern __shared__ __align__(16) char smem[];
  unsigned short* wsm = (unsigned short*)smem;             // 84 tiles * 1KB
  unsigned short* hsm = (unsigned short*)(smem + 86016);   // 32x896 bf16
  float* ghb   = (float*)(smem + 143360);                  // 32x48 f32
  float* hprev = (float*)(smem + 149504);                  // 32x16 f32 (own slice)

  const int c    = blockIdx.x;       // 0..55
  const int tid  = threadIdx.x;
  const int lane = tid & 31;
  const int wave = tid >> 5;         // 0..5

  // Stage this WG's 84 w_hh tiles into LDS (local tile lt = g*28 + kt)
  for (int i = tid; i < 84 * 256; i += 192) {
    int lt = i >> 8, w32 = i & 255;
    int g = lt / 28, kt = lt - g * 28;
    int gtile = (g * NWG_ + c) * 28 + kt;   // global ntile = g*56 + c
    ((unsigned int*)wsm)[i] = ((const unsigned int*)whhp)[(size_t)gtile * 256 + w32];
  }
  for (int i = tid; i < 32 * H_ / 2; i += 192) ((unsigned int*)hsm)[i] = 0u;  // h0 = 0
  for (int i = tid; i < 512; i += 192) hprev[i] = 0.f;
  __syncthreads();

  // Fixed 3 combine slots per thread; slot 2 valid only for tid < 128.
  const bool has2 = (tid + 384) < 512;
  const int eArr[3] = { tid, tid + 192, has2 ? (tid + 384) : 511 };
  float bhr[3], bhz[3], bhn[3];
#pragma unroll
  for (int k = 0; k < 3; ++k) {
    int dd = c * 16 + (eArr[k] & 15);
    bhr[k] = bhh[dd]; bhz[k] = bhh[H_ + dd]; bhn[k] = bhh[2 * H_ + dd];
  }

  const int mt = wave & 1;   // batch half (rows 0..15 / 16..31)
  const int gg = wave >> 1;  // gate 0=r 1=z 2=n
  const unsigned hsm_lds = (unsigned)(unsigned long long)(void*)hsm; // LDS byte addr
  const unsigned long long hbuf64 = (unsigned long long)hbuf;

  for (int t = 0; t < T_; ++t) {
    // --- issue gx loads early: they overlap the gh GEMM below ---
    unsigned short gxu[3][3];
#pragma unroll
    for (int k = 0; k < 3; ++k) {
      int e = eArr[k]; int b_ = e >> 4, dd = c * 16 + (e & 15);
      const unsigned short* gp = gates + ((size_t)t * 32 + b_) * (size_t)G3_ + dd;
      gxu[k][0] = gp[0]; gxu[k][1] = gp[H_]; gxu[k][2] = gp[2 * H_];
    }

    // --- gh = h @ w_hh_slice^T : one 16x16 C tile per wave, K = 896 ---
    v8f acc = {};
#pragma unroll 4
    for (int kt = 0; kt < 28; ++kt) {
      Frag a, b;
      int m = mt * 16 + (lane & 15);
      int byteA = m * (H_ * 2) + kt * 64 + ((lane >> 4) << 5);
      a.q[0] = *reinterpret_cast<const u32x4*>((char*)hsm + byteA);
      a.q[1] = *reinterpret_cast<const u32x4*>((char*)hsm + byteA + 16);
      int byteB = (gg * 28 + kt) * 1024 + lane * 32;
      b.q[0] = *reinterpret_cast<const u32x4*>((char*)wsm + byteB);
      b.q[1] = *reinterpret_cast<const u32x4*>((char*)wsm + byteB + 16);
      acc = __builtin_amdgcn_wmma_f32_16x16x32_bf16(
          false, a.v, false, b.v, (short)0, acc, false, false);
    }
#pragma unroll
    for (int v = 0; v < 8; ++v) {
      int row = mt * 16 + ((lane >> 4) << 3) + v;
      ghb[row * 48 + gg * 16 + (lane & 15)] = acc[v];
    }
    __syncthreads();

    // --- gate nonlinearity + h update (straight-line, predicated slot 2) ---
#pragma unroll
    for (int k = 0; k < 3; ++k) {
      const bool valid = (k < 2) || has2;
      int e = eArr[k]; int b_ = e >> 4, d = e & 15;
      int dd = c * 16 + d;
      float ghr = ghb[b_ * 48 + d]      + bhr[k];
      float ghz = ghb[b_ * 48 + 16 + d] + bhz[k];
      float ghn = ghb[b_ * 48 + 32 + d] + bhn[k];
      float r = fast_sigmoid(bf2f(gxu[k][0]) + ghr);
      float z = fast_sigmoid(bf2f(gxu[k][1]) + ghz);
      float nn = fast_tanh(bf2f(gxu[k][2]) + r * ghn);
      float hp = hprev[e];
      float hn = (1.f - z) * nn + z * hp;
      if (valid) {
        hprev[e] = hn;
        unsigned short hb = f2bf(hn);
        hbuf[b_ * H_ + dd] = hb;                          // broadcast slice
        ornn[((size_t)t * 32 + b_) * H_ + dd] = hb;       // GRU output
      }
    }
    __threadfence();
    grid_sync(bar);   // all slices of h(t) published

    // --- async-pull full h(t) (32x896 bf16 = 57KB) into LDS (ASYNCcnt) ---
    for (int i = tid; i < 3584; i += 192) {               // 3584 x b128
      unsigned lds = hsm_lds + (unsigned)(i * 16);
      unsigned goff = (unsigned)(i * 16);
      asm volatile("global_load_async_to_lds_b128 %0, %1, %2"
                   :: "v"(lds), "v"(goff), "s"(hbuf64) : "memory");
    }
    asm volatile("s_wait_asynccnt 0x0" ::: "memory");
    grid_sync(bar);   // nobody overwrites hbuf until all have copied
  }
}

// ---------------------------------------------------------------------------
extern "C" void kernel_launch(void* const* d_in, const int* in_sizes, int n_in,
                              void* d_out, int out_size, void* d_ws, size_t ws_size,
                              hipStream_t stream) {
  (void)in_sizes; (void)n_in; (void)out_size; (void)ws_size;
  const int*   idx   = (const int*)d_in[0];
  const float* cnd   = (const float*)d_in[1];
  const float* emb   = (const float*)d_in[2];
  const float* w_ih  = (const float*)d_in[3];
  const float* w_hh  = (const float*)d_in[4];
  const float* b_ih  = (const float*)d_in[5];
  const float* b_hh  = (const float*)d_in[6];
  const float* fc1_w = (const float*)d_in[7];
  const float* fc1_b = (const float*)d_in[8];
  const float* fc2_w = (const float*)d_in[9];
  const float* fc2_b = (const float*)d_in[10];
  float* out = (float*)d_out;

  char* ws = (char*)d_ws;
  size_t off = 0;
  auto alloc = [&](size_t bytes) -> char* {
    off = (off + 255) & ~(size_t)255;
    char* p = ws + off; off += bytes; return p;
  };
  unsigned short* wihp  = (unsigned short*)alloc((size_t)168 * 24 * 512 * 2);
  unsigned short* whhp  = (unsigned short*)alloc((size_t)168 * 28 * 512 * 2);
  unsigned short* fc1p  = (unsigned short*)alloc((size_t)56  * 28 * 512 * 2);
  unsigned short* fc2p  = (unsigned short*)alloc((size_t)32  * 28 * 512 * 2);
  unsigned short* xb    = (unsigned short*)alloc((size_t)131072 * 768 * 2);
  unsigned short* gates = (unsigned short*)alloc((size_t)131072 * 2688 * 2);
  unsigned short* ornn  = (unsigned short*)alloc((size_t)131072 * 896 * 2);
  unsigned short* hbuf  = (unsigned short*)alloc((size_t)32 * 896 * 2);
  int* bar = (int*)alloc(256);
  unsigned short* hidden = gates;   // gates dead after the scan; reuse

  { int total = 168 * 24 * 512;
    pack_w<<<(total + 255) / 256, 256, 0, stream>>>(w_ih, wihp, G3_, DIN_, total); }
  { int total = 168 * 28 * 512;
    pack_w<<<(total + 255) / 256, 256, 0, stream>>>(w_hh, whhp, G3_, H_, total); }
  { int total = 56 * 28 * 512;
    pack_w<<<(total + 255) / 256, 256, 0, stream>>>(fc1_w, fc1p, H_, H_, total); }
  { int total = 32 * 28 * 512;
    pack_w<<<(total + 255) / 256, 256, 0, stream>>>(fc2_w, fc2p, O_, H_, total); }

  zero_init<<<(28672 + 255) / 256, 256, 0, stream>>>(hbuf, 28672, bar, 8);

  { int total = 131072 * DIN_;
    gather_xb<<<(total + 255) / 256, 256, 0, stream>>>(idx, cnd, emb, xb, total); }

  // gates_x = x @ w_ih^T + b_ih  (131072 x 2688, K=768)
  { dim3 g(1024, 21);
    gemm_bf16<0, unsigned short><<<g, 256, 0, stream>>>(xb, wihp, b_ih, gates,
                                                        131072, G3_, DIN_); }
  // sequential GRU scan
  gru_scan<<<NWG_, 192, 151552, stream>>>(whhp, gates, b_hh, hbuf, ornn, bar);

  // hidden = relu(o_rnn @ fc1_w^T + b)  (131072 x 896, K=896)
  { dim3 g(1024, 7);
    gemm_bf16<1, unsigned short><<<g, 256, 0, stream>>>(ornn, fc1p, fc1_b, hidden,
                                                        131072, H_, H_); }
  // out = hidden @ fc2_w^T + b, permuted to (B,T,O) f32
  { dim3 g(1024, 4);
    gemm_bf16<2, float><<<g, 256, 0, stream>>>(hidden, fc2p, fc2_b, out,
                                               131072, O_, H_); }
}
